// PhaseNetLoss_74517682586316
// MI455X (gfx1250) — compile-verified, matrix operand order checked by reading, not run on hardware
//
#include <hip/hip_runtime.h>

typedef __attribute__((ext_vector_type(2)))  float    v2f;
typedef __attribute__((ext_vector_type(8)))  float    v8f;
typedef __attribute__((ext_vector_type(16))) _Float16 v16h;
typedef __attribute__((ext_vector_type(8)))  _Float16 v8h;

#define ALPHA 1.0f
#define BETA  0.5f
#define GAMMA 0.3f
#define EPSF  1e-8f

constexpr int Bc = 32, Cc = 256, Fc = 2048;
constexpr int NTRI = 16 * 17 / 2;          // 136 upper-triangular 16x16 tiles

// compile-time triangular tile table: flat t -> (ti, tj), ti <= tj
struct TriTab { unsigned char ti[NTRI]; unsigned char tj[NTRI]; };
constexpr TriTab makeTriTab() {
    TriTab t{};
    int idx = 0;
    for (int i = 0; i < 16; ++i)
        for (int j = i; j < 16; ++j) { t.ti[idx] = (unsigned char)i; t.tj[idx] = (unsigned char)j; ++idx; }
    return t;
}
__device__ __constant__ TriTab TRI = makeTriTab();

// ---------------------------------------------------------------- init
__global__ void init_ws_kernel(float* ws) {
    if (threadIdx.x < 3) ws[threadIdx.x] = 0.0f;
}

// --------------------------- mag + phase loss (streaming) + optional c/s store
template <bool WRITE_CS>
__global__ __launch_bounds__(256) void elem_loss_kernel(
    const float* __restrict__ mh, const float* __restrict__ ph,
    const float* __restrict__ mt, const float* __restrict__ pt,
    _Float16* __restrict__ ch, _Float16* __restrict__ sh,
    float* __restrict__ ws, int n) {
    float magAcc = 0.0f, phAcc = 0.0f;
    for (int i = blockIdx.x * blockDim.x + threadIdx.x; i < n;
         i += gridDim.x * blockDim.x) {
        // pure streaming operands: non-temporal so they don't evict L2-resident c/s
        float dm = __builtin_nontemporal_load(mh + i) -
                   __builtin_nontemporal_load(mt + i);
        magAcc += dm * dm;
        float p = ph[i];
        phAcc  += 1.0f - __cosf(p - __builtin_nontemporal_load(pt + i));
        if (WRITE_CS) {
            ch[i] = (_Float16)__cosf(p);   // stays cached (RT) for the GEMM pass
            sh[i] = (_Float16)__sinf(p);
        }
    }
    for (int off = 16; off > 0; off >>= 1) {
        magAcc += __shfl_down(magAcc, off, 32);
        phAcc  += __shfl_down(phAcc,  off, 32);
    }
    __shared__ float sm[8], sp[8];
    const int wave = threadIdx.x >> 5;
    const int lane = threadIdx.x & 31;
    if (lane == 0) { sm[wave] = magAcc; sp[wave] = phAcc; }
    __syncthreads();
    if (threadIdx.x == 0) {
        float m = 0.0f, p = 0.0f;
        for (int w = 0; w < 8; ++w) { m += sm[w]; p += sp[w]; }
        atomicAdd(ws + 0, m);
        atomicAdd(ws + 1, p);
    }
}

// --------------------------------- PCI Gram matrices via f16 WMMA (fast path)
// One wave = one upper-triangular 16x16 output tile. pci_hat is exactly
// symmetric (real sym, imag antisym, same summation order), so off-diagonal
// tiles score both (i,j) and (j,i) against pci_target from one WMMA result.
// 16-bit A/B layout (ISA 7.12.2): row/col = lane&15, half h = lane>>4; per lane
// K = [8h,8h+8) in VGPR0-3 and [16+8h,24+8h) in VGPR4-7 -> two b128 loads/frag.
__global__ __launch_bounds__(256) void pci_loss_f16_kernel(
    const _Float16* __restrict__ ch, const _Float16* __restrict__ sh,
    const float* __restrict__ pci_t, float* __restrict__ ws) {
    const int lane = threadIdx.x & 31;
    const int wave = threadIdx.x >> 5;

    const int tile = blockIdx.x * 8 + wave;       // 0 .. 32*136-1
    const int b  = tile / NTRI;
    const int t  = tile - b * NTRI;
    const int ti = TRI.ti[t];
    const int tj = TRI.tj[t];

    const int row  = lane & 15;
    const int hsel = lane >> 4;
    const int o0 = 8 * hsel;
    const int o1 = 16 + 8 * hsel;

    const size_t aOff = ((size_t)b * Cc + (ti * 16 + row)) * Fc;
    const size_t bOff = ((size_t)b * Cc + (tj * 16 + row)) * Fc;
    const _Float16* ca_p = ch + aOff;
    const _Float16* sa_p = sh + aOff;
    const _Float16* cb_p = ch + bOff;
    const _Float16* sb_p = sh + bOff;

    v8f acc_cc = {}, acc_ss = {}, acc_sc = {}, acc_cs = {};

    for (int k = 0; k < Fc; k += 32) {
        union { v16h v; v8h h[2]; } fca, fsa, fcb, fsb;
        fca.h[0] = *(const v8h*)(ca_p + k + o0);
        fca.h[1] = *(const v8h*)(ca_p + k + o1);
        fsa.h[0] = *(const v8h*)(sa_p + k + o0);
        fsa.h[1] = *(const v8h*)(sa_p + k + o1);
        fcb.h[0] = *(const v8h*)(cb_p + k + o0);
        fcb.h[1] = *(const v8h*)(cb_p + k + o1);
        fsb.h[0] = *(const v8h*)(sb_p + k + o0);
        fsb.h[1] = *(const v8h*)(sb_p + k + o1);
        acc_cc = __builtin_amdgcn_wmma_f32_16x16x32_f16(false, fca.v, false, fcb.v,
                                                        (short)0, acc_cc, false, false);
        acc_ss = __builtin_amdgcn_wmma_f32_16x16x32_f16(false, fsa.v, false, fsb.v,
                                                        (short)0, acc_ss, false, false);
        acc_sc = __builtin_amdgcn_wmma_f32_16x16x32_f16(false, fsa.v, false, fcb.v,
                                                        (short)0, acc_sc, false, false);
        acc_cs = __builtin_amdgcn_wmma_f32_16x16x32_f16(false, fca.v, false, fsb.v,
                                                        (short)0, acc_cs, false, false);
    }

    const float invF  = 1.0f / (float)Fc;
    const int   iBase = ti * 16 + (hsel << 3);
    const int   j     = tj * 16 + row;
    const bool  offDiag = (ti != tj);
    float local = 0.0f;
#pragma unroll
    for (int r = 0; r < 8; ++r) {
        const int i = iBase + r;
        float real = (acc_cc[r] + acc_ss[r]) * invF;
        float imag = (acc_sc[r] - acc_cs[r]) * invF;
        float pci  = sqrtf(real * real + imag * imag + EPSF);
        float d    = pci - pci_t[((size_t)b * Cc + i) * Cc + j];
        local += d * d;
        if (offDiag) {                      // mirrored element, same pci value
            float d2 = pci - pci_t[((size_t)b * Cc + j) * Cc + i];
            local += d2 * d2;
        }
    }
    for (int off = 16; off > 0; off >>= 1) local += __shfl_down(local, off, 32);
    if (lane == 0) atomicAdd(ws + 2, local);
}

// ------------------- fallback: f32 WMMA with in-tile sincos (no big workspace)
__global__ __launch_bounds__(256) void pci_loss_f32_kernel(
    const float* __restrict__ phase, const float* __restrict__ pci_t,
    float* __restrict__ ws) {
    const int lane = threadIdx.x & 31;
    const int wave = threadIdx.x >> 5;

    const int tile = blockIdx.x * 8 + wave;       // 0 .. 32*136-1
    const int b  = tile / NTRI;
    const int t  = tile - b * NTRI;
    const int ti = TRI.ti[t];
    const int tj = TRI.tj[t];

    const int row  = lane & 15;
    const int koff = (lane >> 4) << 1;

    const float* pa = phase + ((size_t)b * Cc + (ti * 16 + row)) * Fc + koff;
    const float* pb = phase + ((size_t)b * Cc + (tj * 16 + row)) * Fc + koff;

    v8f acc_cc = {}, acc_ss = {}, acc_sc = {}, acc_cs = {};
    for (int k = 0; k < Fc; k += 4) {
        v2f A  = *(const v2f*)(pa + k);
        v2f Bv = *(const v2f*)(pb + k);
        v2f ca, sa, cb, sb;
        ca.x = __cosf(A.x);  ca.y = __cosf(A.y);
        sa.x = __sinf(A.x);  sa.y = __sinf(A.y);
        cb.x = __cosf(Bv.x); cb.y = __cosf(Bv.y);
        sb.x = __sinf(Bv.x); sb.y = __sinf(Bv.y);
        acc_cc = __builtin_amdgcn_wmma_f32_16x16x4_f32(false, ca, false, cb,
                                                       (short)0, acc_cc, false, false);
        acc_ss = __builtin_amdgcn_wmma_f32_16x16x4_f32(false, sa, false, sb,
                                                       (short)0, acc_ss, false, false);
        acc_sc = __builtin_amdgcn_wmma_f32_16x16x4_f32(false, sa, false, cb,
                                                       (short)0, acc_sc, false, false);
        acc_cs = __builtin_amdgcn_wmma_f32_16x16x4_f32(false, ca, false, sb,
                                                       (short)0, acc_cs, false, false);
    }

    const float invF  = 1.0f / (float)Fc;
    const int   iBase = ti * 16 + ((lane >> 4) << 3);
    const int   j     = tj * 16 + row;
    const bool  offDiag = (ti != tj);
    float local = 0.0f;
#pragma unroll
    for (int r = 0; r < 8; ++r) {
        const int i = iBase + r;
        float real = (acc_cc[r] + acc_ss[r]) * invF;
        float imag = (acc_sc[r] - acc_cs[r]) * invF;
        float pci  = sqrtf(real * real + imag * imag + EPSF);
        float d    = pci - pci_t[((size_t)b * Cc + i) * Cc + j];
        local += d * d;
        if (offDiag) {
            float d2 = pci - pci_t[((size_t)b * Cc + j) * Cc + i];
            local += d2 * d2;
        }
    }
    for (int off = 16; off > 0; off >>= 1) local += __shfl_down(local, off, 32);
    if (lane == 0) atomicAdd(ws + 2, local);
}

// ---------------------------------------------------------------- finalize
__global__ void finalize_kernel(const float* __restrict__ ws,
                                float* __restrict__ out) {
    if (threadIdx.x == 0) {
        const float nElem = (float)Bc * (float)Cc * (float)Fc;
        const float nPci  = (float)Bc * (float)Cc * (float)Cc;
        float mag = ws[0] / nElem;
        float ph  = ws[1] / nElem;
        float coh = ws[2] / nPci;
        out[0] = ALPHA * mag + BETA * ph + GAMMA * coh;
        out[1] = mag;
        out[2] = ph;
        out[3] = coh;
    }
}

extern "C" void kernel_launch(void* const* d_in, const int* in_sizes, int n_in,
                              void* d_out, int out_size, void* d_ws, size_t ws_size,
                              hipStream_t stream) {
    const float* mh    = (const float*)d_in[0];
    const float* ph    = (const float*)d_in[1];
    const float* mt    = (const float*)d_in[2];
    const float* pt    = (const float*)d_in[3];
    const float* pci_t = (const float*)d_in[4];
    float* ws  = (float*)d_ws;
    float* out = (float*)d_out;
    const int n = in_sizes[0];                     // B*C*F = 16,777,216

    const size_t csBytes = (size_t)n * sizeof(_Float16);
    const size_t need    = 256 + 2 * csBytes;
    _Float16* ch = (_Float16*)((char*)d_ws + 256);
    _Float16* sh = (_Float16*)((char*)d_ws + 256 + csBytes);

    const int triBlocks = (Bc * NTRI) / 8;         // 4352 waves / 8 = 544 blocks

    init_ws_kernel<<<1, 32, 0, stream>>>(ws);
    if (ws_size >= need) {
        // fast path: precompute cos/sin once (f16), Gram via v_wmma_f32_16x16x32_f16
        elem_loss_kernel<true><<<4096, 256, 0, stream>>>(mh, ph, mt, pt, ch, sh, ws, n);
        pci_loss_f16_kernel<<<triBlocks, 256, 0, stream>>>(ch, sh, pci_t, ws);
    } else {
        // fallback: recompute sincos per tile, Gram via v_wmma_f32_16x16x4_f32
        elem_loss_kernel<false><<<4096, 256, 0, stream>>>(mh, ph, mt, pt, nullptr, nullptr, ws, n);
        pci_loss_f32_kernel<<<triBlocks, 256, 0, stream>>>(ph, pci_t, ws);
    }
    finalize_kernel<<<1, 32, 0, stream>>>(ws, out);
}
